// SelfAttention_90666759618654
// MI455X (gfx1250) — matrix operand order repaired and checked by reading, not compiled, not run on hardware
//
#include <hip/hip_runtime.h>

#define D_MODEL 1024
#define D_OUT   64
#define BATCH   4
#define SEQ     4096

typedef __attribute__((ext_vector_type(16))) _Float16 v16h;
typedef __attribute__((ext_vector_type(8)))  _Float16 v8h;
typedef __attribute__((ext_vector_type(4)))  _Float16 v4h;
typedef __attribute__((ext_vector_type(8)))  float    v8f;
typedef __attribute__((ext_vector_type(4)))  float    v4f;

// Load 8 contiguous fp32 and convert to 8 halfs.
__device__ __forceinline__ v8h cvt8(const float* p) {
    v4f a = *(const v4f*)p;
    v4f b = *(const v4f*)(p + 4);
    v4h ha = __builtin_convertvector(a, v4h);
    v4h hb = __builtin_convertvector(b, v4h);
    return __builtin_shufflevector(ha, hb, 0,1,2,3,4,5,6,7);
}
__device__ __forceinline__ v16h cat16(v8h lo, v8h hi) {
    return __builtin_shufflevector(lo, hi, 0,1,2,3,4,5,6,7,8,9,10,11,12,13,14,15);
}
// B-fragment: this lane needs 16 contiguous halfs along K.
__device__ __forceinline__ v16h loadB(const _Float16* p) {
    v8h lo = *(const v8h*)p;
    v8h hi = *(const v8h*)(p + 8);
    return cat16(lo, hi);
}

#define WMMA_F16(A, B, C) \
    __builtin_amdgcn_wmma_f32_16x16x32_f16(false, (A), false, (B), (short)0, (C), false, false)

// ---------------- Stage 0: transpose + f16-convert the three weight matrices
// W[p] is [1024, 64] fp32 -> wT[p] is [64, 1024] f16 (K contiguous per out-feature).
__global__ __launch_bounds__(256)
void wt_kernel(const float* __restrict__ Wq, const float* __restrict__ Wk,
               const float* __restrict__ Wv, _Float16* __restrict__ wT) {
    int i = blockIdx.x * blockDim.x + threadIdx.x;     // 0 .. 3*64*1024-1
    int p = i / (D_OUT * D_MODEL);
    int r = i % (D_OUT * D_MODEL);
    int e = r / D_MODEL;
    int d = r % D_MODEL;
    const float* W = (p == 0) ? Wq : ((p == 1) ? Wk : Wv);
    wT[i] = (_Float16)W[d * D_OUT + e];
}

// ---------------- Stage 1: projection  out[row, 0..63] = (X[row,:] @ W + b) * out_scale
// One wave per 16-row tile; K loop unrolled x2 with ping-pong fragment buffers
// (no register rotation movs). 4 waves / block.
__global__ __launch_bounds__(128)
void proj_kernel(const float* __restrict__ X, const _Float16* __restrict__ wT,
                 const float* __restrict__ bias,
                 _Float16* __restrict__ outRM,   // row-major [B*S, 64]   (q, k)
                 _Float16* __restrict__ outT,    // transposed [B][64][S] (v)
                 int transpose_out, float out_scale) {
    int wid  = threadIdx.x >> 5;
    int lane = threadIdx.x & 31;
    int tile = blockIdx.x * 4 + wid;     // 0..1023
    int row0 = tile * 16;                // flat row over B*S
    int n    = lane & 15;
    int hi   = lane >> 4;
    int abase = hi * 16;                 // A-frag K sub-base
    int kbase = hi * 16;                 // B-frag K sub-base

    const float* xrow = X + (size_t)(row0 + n) * D_MODEL; // A row = row0 + lane%16
    const _Float16* wrow[4];
    #pragma unroll
    for (int t = 0; t < 4; ++t)
        wrow[t] = wT + (size_t)(t * 16 + n) * D_MODEL + kbase;

    v8f acc[4] = {};

    // phase-0 buffers preloaded for k=0
    v16h a0 = cat16(cvt8(xrow + abase), cvt8(xrow + 8 + abase));
    v16h b0[4];
    #pragma unroll
    for (int t = 0; t < 4; ++t) b0[t] = loadB(wrow[t]);

    for (int k = 0; k < D_MODEL; k += 64) {
        __builtin_prefetch(xrow + k + 192, 0, 1);       // stream X ahead
        // preload phase-1 (k+32) while phase-0 WMMAs run
        v16h a1 = cat16(cvt8(xrow + k + 32 + abase), cvt8(xrow + k + 40 + abase));
        v16h b1[4];
        #pragma unroll
        for (int t = 0; t < 4; ++t) b1[t] = loadB(wrow[t] + k + 32);
        #pragma unroll
        for (int t = 0; t < 4; ++t) acc[t] = WMMA_F16(a0, b0[t], acc[t]);

        // preload phase-0 for k+64 (dummy reload of k=0 on last iter)
        int kn = (k + 64 < D_MODEL) ? (k + 64) : 0;
        a0 = cat16(cvt8(xrow + kn + abase), cvt8(xrow + kn + 8 + abase));
        #pragma unroll
        for (int t = 0; t < 4; ++t) b0[t] = loadB(wrow[t] + kn);
        #pragma unroll
        for (int t = 0; t < 4; ++t) acc[t] = WMMA_F16(a1, b1[t], acc[t]);
    }

    float bl[4];
    #pragma unroll
    for (int t = 0; t < 4; ++t) bl[t] = bias[t * 16 + n];

    int b  = row0 / SEQ;
    int s0 = row0 % SEQ;
    #pragma unroll
    for (int t = 0; t < 4; ++t) {
        #pragma unroll
        for (int r = 0; r < 8; ++r) {
            float val = (acc[t][r] + bl[t]) * out_scale;
            int rr = r + hi * 8;
            if (!transpose_out) {
                outRM[(size_t)(row0 + rr) * D_OUT + t * 16 + n] = (_Float16)val;
            } else {
                outT[((size_t)b * D_OUT + t * 16 + n) * SEQ + (s0 + rr)] = (_Float16)val;
            }
        }
    }
}

// ---------------- Stage 2: split-K flash attention
// One block (4 waves) per 16-query tile; wave w covers keys [w*1024, (w+1)*1024),
// partial (m, l, acc) merged through LDS at the end. K fragments are
// double-buffered across iterations so QK WMMAs never wait on fresh loads.
__global__ __launch_bounds__(128)
void flash_kernel(const _Float16* __restrict__ qf,   // [B*S, 64] row-major (pre-scaled by 1/8)
                  const _Float16* __restrict__ kf,   // [B*S, 64] row-major
                  const _Float16* __restrict__ vT,   // [B][64][S]
                  float* __restrict__ out) {         // [B*S, 64] fp32
    __shared__ __attribute__((aligned(16))) _Float16 lds_p[4][16 * 48]; // per-wave P tile
    __shared__ float lds_acc[4][16][D_OUT];                             // partial outputs
    __shared__ float lds_m[4][16];
    __shared__ float lds_l[4][16];

    int wid  = threadIdx.x >> 5;
    int lane = threadIdx.x & 31;
    int tile = blockIdx.x;                    // 0..1023
    int b    = tile / (SEQ / 16);
    int q0   = (tile % (SEQ / 16)) * 16;
    int n    = lane & 15;
    int hi   = lane >> 4;
    int abase = hi * 16;
    int kbase = hi * 16;
    const int KSPAN = SEQ / 4;                // keys per wave
    int j0 = wid * KSPAN;

    const _Float16* qrow = qf + (size_t)(b * SEQ + q0 + n) * D_OUT;
    v16h aq0 = cat16(*(const v8h*)(qrow + abase),      *(const v8h*)(qrow + 8 + abase));
    v16h aq1 = cat16(*(const v8h*)(qrow + 32 + abase), *(const v8h*)(qrow + 40 + abase));

    const _Float16* kbat = kf + (size_t)b * SEQ * D_OUT;
    const _Float16* vbat = vT + (size_t)b * D_OUT * SEQ;
    _Float16* myp = &lds_p[wid][0];

    v8f acc[4] = {};
    float m[8], l[8];
    #pragma unroll
    for (int r = 0; r < 8; ++r) { m[r] = -1e30f; l[r] = 0.0f; }

    // preload K fragments for first block
    v16h kb[4];
    {
        const _Float16* k0 = kbat + (size_t)(j0 + n) * D_OUT + kbase;
        const _Float16* k1 = kbat + (size_t)(j0 + 16 + n) * D_OUT + kbase;
        kb[0] = loadB(k0);  kb[1] = loadB(k0 + 32);
        kb[2] = loadB(k1);  kb[3] = loadB(k1 + 32);
    }

    for (int j = j0; j < j0 + KSPAN; j += 32) {
        // ---- V fragments issued early: consumed only after softmax ----
        v16h vb[4];
        #pragma unroll
        for (int t = 0; t < 4; ++t)
            vb[t] = loadB(vbat + (size_t)(t * 16 + n) * SEQ + j + kbase);

        // ---- S = Q @ K^T for 32 keys using pre-loaded fragments ----
        v8f c0 = {}, c1 = {};
        c0 = WMMA_F16(aq0, kb[0], c0);
        c0 = WMMA_F16(aq1, kb[1], c0);
        c1 = WMMA_F16(aq0, kb[2], c1);
        c1 = WMMA_F16(aq1, kb[3], c1);

        // reload K fragments for j+32 (dummy reload of j0 on last iter);
        // these loads complete during softmax + LDS round-trip
        int jn = (j + 32 < j0 + KSPAN) ? (j + 32) : j0;
        const _Float16* k0 = kbat + (size_t)(jn + n) * D_OUT + kbase;
        const _Float16* k1 = kbat + (size_t)(jn + 16 + n) * D_OUT + kbase;
        kb[0] = loadB(k0);  kb[1] = loadB(k0 + 32);
        kb[2] = loadB(k1);  kb[3] = loadB(k1 + 32);

        // warm the cache two blocks ahead
        __builtin_prefetch(kbat + (size_t)(j + 64 + n) * D_OUT, 0, 1);
        __builtin_prefetch(kbat + (size_t)(j + 80 + n) * D_OUT, 0, 1);

        // ---- online softmax (row = r + hi*8, cols across 16 lanes) ----
        float p0[8], p1[8], alpha[8];
        #pragma unroll
        for (int r = 0; r < 8; ++r) {
            float s0v = c0[r];
            float s1v = c1[r];
            float mx = fmaxf(s0v, s1v);
            #pragma unroll
            for (int msk = 1; msk < 16; msk <<= 1)
                mx = fmaxf(mx, __shfl_xor(mx, msk, 32));
            float mn = fmaxf(m[r], mx);
            alpha[r] = __expf(m[r] - mn);
            m[r] = mn;
            p0[r] = __expf(s0v - mn);
            p1[r] = __expf(s1v - mn);
            float rs = p0[r] + p1[r];
            #pragma unroll
            for (int msk = 1; msk < 16; msk <<= 1)
                rs += __shfl_xor(rs, msk, 32);
            l[r] = l[r] * alpha[r] + rs;
        }
        #pragma unroll
        for (int t = 0; t < 4; ++t)
            #pragma unroll
            for (int r = 0; r < 8; ++r)
                acc[t][r] *= alpha[r];

        // ---- wave-local relayout of P (C-layout -> A-fragment) via LDS ----
        #pragma unroll
        for (int r = 0; r < 8; ++r) {
            int row = r + hi * 8;
            myp[row * 48 + n]      = (_Float16)p0[r];
            myp[row * 48 + 16 + n] = (_Float16)p1[r];
        }
        asm volatile("s_wait_dscnt 0" ::: "memory");
        __builtin_amdgcn_wave_barrier();
        v16h ap = cat16(*(const v8h*)(myp + n * 48 + abase),
                        *(const v8h*)(myp + n * 48 + 8 + abase));
        asm volatile("" ::: "memory");

        // ---- O += P @ V_block ----
        #pragma unroll
        for (int t = 0; t < 4; ++t)
            acc[t] = WMMA_F16(ap, vb[t], acc[t]);
    }

    // ---- publish partials ----
    #pragma unroll
    for (int t = 0; t < 4; ++t)
        #pragma unroll
        for (int r = 0; r < 8; ++r)
            lds_acc[wid][r + hi * 8][t * 16 + n] = acc[t][r];
    if (n == 0) {
        #pragma unroll
        for (int r = 0; r < 8; ++r) {
            lds_m[wid][r + hi * 8] = m[r];
            lds_l[wid][r + hi * 8] = l[r];
        }
    }
    __syncthreads();

    // ---- merge: wave `wid` finalizes feature tile t = wid ----
    int t = wid;
    #pragma unroll
    for (int r = 0; r < 8; ++r) {
        int row = r + hi * 8;
        float m0 = lds_m[0][row], m1 = lds_m[1][row];
        float m2 = lds_m[2][row], m3 = lds_m[3][row];
        float M = fmaxf(fmaxf(m0, m1), fmaxf(m2, m3));
        float e0 = __expf(m0 - M), e1 = __expf(m1 - M);
        float e2 = __expf(m2 - M), e3 = __expf(m3 - M);
        float L = lds_l[0][row] * e0 + lds_l[1][row] * e1 +
                  lds_l[2][row] * e2 + lds_l[3][row] * e3;
        int col = t * 16 + n;
        float s = lds_acc[0][row][col] * e0 + lds_acc[1][row][col] * e1 +
                  lds_acc[2][row][col] * e2 + lds_acc[3][row][col] * e3;
        out[(size_t)(b * SEQ + q0 + row) * D_OUT + col] = s / L;
    }
}

extern "C" void kernel_launch(void* const* d_in, const int* in_sizes, int n_in,
                              void* d_out, int out_size, void* d_ws, size_t ws_size,
                              hipStream_t stream) {
    const float* query = (const float*)d_in[0];
    const float* key   = (const float*)d_in[1];
    const float* value = (const float*)d_in[2];
    const float* Wq    = (const float*)d_in[3];
    const float* bq    = (const float*)d_in[4];
    const float* Wk    = (const float*)d_in[5];
    const float* bk    = (const float*)d_in[6];
    const float* Wv    = (const float*)d_in[7];
    const float* bv    = (const float*)d_in[8];

    const size_t NPROJ = (size_t)BATCH * SEQ * D_OUT;    // 1,048,576 halfs each
    _Float16* qf = (_Float16*)d_ws;
    _Float16* kf = qf + NPROJ;
    _Float16* vT = kf + NPROJ;
    _Float16* wT = vT + NPROJ;                           // 3 * 64 * 1024 halfs

    wt_kernel<<<(3 * D_OUT * D_MODEL) / 256, 256, 0, stream>>>(Wq, Wk, Wv, wT);

    // 1024 sixteen-row tiles, 4 waves per 128-thread block -> 256 blocks
    // q is pre-scaled by 1/sqrt(64) so the attention loop skips the scale.
    proj_kernel<<<256, 128, 0, stream>>>(query, wT,                       bq, qf, nullptr, 0, 0.125f);
    proj_kernel<<<256, 128, 0, stream>>>(key,   wT + D_OUT * D_MODEL,     bk, kf, nullptr, 0, 1.0f);
    proj_kernel<<<256, 128, 0, stream>>>(value, wT + 2 * D_OUT * D_MODEL, bv, nullptr, vT, 1, 1.0f);

    // one block (4 split-K waves) per 16-query tile
    flash_kernel<<<1024, 128, 0, stream>>>(qf, kf, vT, (float*)d_out);
}
